// GPModel_17841294147860
// MI455X (gfx1250) — compile-verified, matrix operand order checked by reading, not run on hardware
//
#include <hip/hip_runtime.h>
#include <hip/hip_bf16.h>

// ---------------------------------------------------------------------------
// MI455X (gfx1250) implementation.
// Shapes: B=16, L=1024, H=1024, T=2. Output (16,2,1024,1024) fp32.
// Strategy: bf16 WMMA (v_wmma_f32_16x16x32_bf16) for both big GEMMs with fp32
// accumulation; software-pipelined K-loops (prefetch next fragments before
// issuing current WMMAs) to avoid s_wait_loadcnt 0 before every matrix op;
// fused RoPE epilogue on GEMM1; fused bias/mask epilogue on batched QK^T.
// All staging buffers (hidden_bf16, W1^T bf16, q, k, rope tables, bias)
// live in d_ws and fit in the 192MB L2.
// ---------------------------------------------------------------------------

typedef __attribute__((ext_vector_type(16))) __bf16 v16bf;
typedef __attribute__((ext_vector_type(8)))  float  v8f;
typedef __attribute__((ext_vector_type(8)))  unsigned short us8;
typedef __attribute__((ext_vector_type(16))) unsigned short us16;

#define NEGBIG (-1000000000000.0f)

__device__ __forceinline__ unsigned short f2bf(float f) {
    unsigned int u = __builtin_bit_cast(unsigned int, f);
    u += 0x7FFFu + ((u >> 16) & 1u);            // round-to-nearest-even
    return (unsigned short)(u >> 16);
}

// Fragment loader per ISA 7.12.2 (16-bit A 16x32 / B 32x16):
// lanes 0-15 hold row/col laneN with K in {0..7, 16..23}; lanes 16-31 the same
// row/col with K in {8..15, 24..31}.  p already includes row*ld + hi*8.
__device__ __forceinline__ v16bf lf(const unsigned short* __restrict__ p, int kk) {
    us8 c0 = *(const us8*)(p + kk);
    us8 c1 = *(const us8*)(p + kk + 16);
    us16 u;
#pragma unroll
    for (int i = 0; i < 8; ++i) { u[i] = c0[i]; u[i + 8] = c1[i]; }
    return __builtin_bit_cast(v16bf, u);
}

// ------------------------------- prep kernels ------------------------------

__global__ void __launch_bounds__(256) k_cvt_hidden(const float* __restrict__ src,
                                                    unsigned short* __restrict__ dst) {
    int idx = blockIdx.x * 256 + threadIdx.x;          // 16777216 elements
    dst[idx] = f2bf(src[idx]);
}

__global__ void __launch_bounds__(256) k_w1t(const float* __restrict__ W1,
                                             unsigned short* __restrict__ w1t) {
    int idx = blockIdx.x * 256 + threadIdx.x;          // 2048*1024
    int n  = idx >> 10;
    int kk = idx & 1023;
    w1t[idx] = f2bf(W1[kk * 2048 + n]);
}

__global__ void __launch_bounds__(256) k_rope_tab(float* __restrict__ cosT,
                                                  float* __restrict__ sinT) {
    int idx = blockIdx.x * 256 + threadIdx.x;          // 1024*512
    int l = idx >> 9;
    int i = idx & 511;
    // inv = 10000^(-2i/H)
    float inv = __expf(-9.2103403719761836f * (float)(2 * i) * (1.0f / 1024.0f));
    float ang = (float)l * inv;
    float s, c;
    __sincosf(ang, &s, &c);
    cosT[idx] = c;
    sinT[idx] = s;
}

// bias2[b][c][l] = (hidden[b,l,:].W2[:,c] + b2[c]) * 0.5 ; c in 0..3
__global__ void __launch_bounds__(256) k_bias2(const float* __restrict__ hidden,
                                               const float* __restrict__ W2,
                                               const float* __restrict__ b2,
                                               float* __restrict__ bias2) {
    int idx = blockIdx.x * 256 + threadIdx.x;          // 16384 = B*L
    int b = idx >> 10;
    int l = idx & 1023;
    const float* hrow = hidden + (size_t)idx * 1024;
    float a0 = 0.f, a1 = 0.f, a2 = 0.f, a3 = 0.f;
    for (int h = 0; h < 1024; ++h) {
        float hv = hrow[h];
        const float* wv = W2 + h * 4;
        a0 += hv * wv[0]; a1 += hv * wv[1]; a2 += hv * wv[2]; a3 += hv * wv[3];
    }
    float* o = bias2 + b * 4096 + l;
    o[0]    = (a0 + b2[0]) * 0.5f;
    o[1024] = (a1 + b2[1]) * 0.5f;
    o[2048] = (a2 + b2[2]) * 0.5f;
    o[3072] = (a3 + b2[3]) * 0.5f;
}

// ------------------------ GEMM1 + RoPE (q,k producer) ----------------------
// O = hidden @ W1 + b1 ; even cols -> q, odd cols -> k ; RoPE pairs columns
// (2i,2i+1) of q (resp. k), which are O-columns 2 apart -> shfl_xor(.,2).
__global__ void __launch_bounds__(256) gemm1_rope(
    const unsigned short* __restrict__ hid,   // 16384 x 1024 bf16
    const unsigned short* __restrict__ w1t,   // 2048 x 1024 bf16 (W1 transposed)
    const float* __restrict__ b1,             // 2048
    const float* __restrict__ cosT,           // 1024 x 512
    const float* __restrict__ sinT,
    unsigned short* __restrict__ qb,          // 16384 x 1024 bf16
    unsigned short* __restrict__ kb_) {
    int w     = blockIdx.x * 8 + (threadIdx.x >> 5);
    int lane  = threadIdx.x & 31;
    int laneN = lane & 15;
    int hi    = lane >> 4;
    int mt = w >> 5;                 // 1024 M-tiles
    int ng = w & 31;                 // 32 groups of 64 N-cols
    int m0  = mt * 16;
    int n0g = ng * 64;

    const unsigned short* pa = hid + (size_t)(m0 + laneN) * 1024 + hi * 8;
    const unsigned short* pb = w1t + (size_t)(n0g + laneN) * 1024 + hi * 8;

    v8f vzero = {0.f,0.f,0.f,0.f,0.f,0.f,0.f,0.f};
    v8f acc[4];
#pragma unroll
    for (int nt = 0; nt < 4; ++nt) acc[nt] = vzero;

    // software pipeline: loads for kk+32 are issued before WMMAs for kk
    v16bf aC = lf(pa, 0);
    v16bf bC[4];
#pragma unroll
    for (int nt = 0; nt < 4; ++nt) bC[nt] = lf(pb + nt * 16384, 0);

#pragma unroll 2
    for (int kk = 0; kk < 1024; kk += 32) {
        int kn = (kk + 32) & 1023;           // wraps to 0 on last iteration
        v16bf aN = lf(pa, kn);
        v16bf bN[4];
#pragma unroll
        for (int nt = 0; nt < 4; ++nt) bN[nt] = lf(pb + nt * 16384, kn);
#pragma unroll
        for (int nt = 0; nt < 4; ++nt)
            acc[nt] = __builtin_amdgcn_wmma_f32_16x16x32_bf16(
                false, aC, false, bC[nt], (short)0, acc[nt], false, false);
        aC = aN;
#pragma unroll
        for (int nt = 0; nt < 4; ++nt) bC[nt] = bN[nt];
    }

#pragma unroll
    for (int nt = 0; nt < 4; ++nt) {
        int colO = n0g + nt * 16 + laneN;
        float bv = b1[colO];
        int i = colO >> 2;                       // rope pair index
        int j = colO >> 1;                       // q/k column index
        unsigned short* dst = (colO & 1) ? kb_ : qb;
        bool second = (laneN & 2) != 0;          // second element of rope pair
#pragma unroll
        for (int r = 0; r < 8; ++r) {
            int m = m0 + r + hi * 8;
            int l = m & 1023;
            float v = acc[nt][r] + bv;
            float p = __shfl_xor(v, 2, 32);      // rope partner column
            float c = cosT[l * 512 + i];
            float s = sinT[l * 512 + i];
            float o = second ? (v * c + p * s) : (v * c - p * s);
            dst[(size_t)m * 1024 + j] = f2bf(o);
        }
    }
}

// -------------------- batched QK^T + bias + masks + store ------------------
__global__ void __launch_bounds__(256) gemm2_out(
    const unsigned short* __restrict__ qb,
    const unsigned short* __restrict__ kbuf,
    const float* __restrict__ bias2,          // [16][4][1024]
    const int* __restrict__ ttid,             // [16][1024]
    float* __restrict__ out) {                // [16][2][1024][1024]
    int w     = blockIdx.x * 8 + (threadIdx.x >> 5);
    int lane  = threadIdx.x & 31;
    int laneN = lane & 15;
    int hi    = lane >> 4;
    int b   = w >> 10;
    int rem = w & 1023;
    int mt  = rem >> 4;                // 64 M-tiles
    int ng  = rem & 15;                // 16 groups of 64 N-cols
    int m0  = mt * 16;
    int n0g = ng * 64;

    const unsigned short* q = qb   + (size_t)b * (1024 * 1024);
    const unsigned short* k = kbuf + (size_t)b * (1024 * 1024);
    const unsigned short* pa = q + (size_t)(m0 + laneN) * 1024 + hi * 8;
    const unsigned short* pb = k + (size_t)(n0g + laneN) * 1024 + hi * 8;

    v8f vzero = {0.f,0.f,0.f,0.f,0.f,0.f,0.f,0.f};
    v8f acc[4];
#pragma unroll
    for (int nt = 0; nt < 4; ++nt) acc[nt] = vzero;

    // software pipeline
    v16bf aC = lf(pa, 0);
    v16bf bC[4];
#pragma unroll
    for (int nt = 0; nt < 4; ++nt) bC[nt] = lf(pb + nt * 16384, 0);

#pragma unroll 2
    for (int kk = 0; kk < 1024; kk += 32) {
        int kn = (kk + 32) & 1023;
        v16bf aN = lf(pa, kn);
        v16bf bN[4];
#pragma unroll
        for (int nt = 0; nt < 4; ++nt) bN[nt] = lf(pb + nt * 16384, kn);
#pragma unroll
        for (int nt = 0; nt < 4; ++nt)
            acc[nt] = __builtin_amdgcn_wmma_f32_16x16x32_bf16(
                false, aC, false, bC[nt], (short)0, acc[nt], false, false);
        aC = aN;
#pragma unroll
        for (int nt = 0; nt < 4; ++nt) bC[nt] = bN[nt];
    }

    const float* bb = bias2 + b * 4096;
    const int*   tt = ttid + b * 1024;
#pragma unroll
    for (int nt = 0; nt < 4; ++nt) {
        int n = n0g + nt * 16 + laneN;
        float bn0 = bb[n];                 // channel 0 -> t=0, broadcast over m
        float bn1 = bb[2048 + n];          // channel 2 -> t=1
        bool  cm  = (tt[n] == 0);
        float* o0 = out + (((size_t)(b * 2 + 0)) << 20) + n;
        float* o1 = out + (((size_t)(b * 2 + 1)) << 20) + n;
#pragma unroll
        for (int r = 0; r < 8; ++r) {
            int m = m0 + r + hi * 8;
            float s   = acc[nt][r] * 0.03125f;        // 1/sqrt(1024)
            float bm0 = bb[1024 + m];                 // channel 1 -> t=0, over n
            float bm1 = bb[3072 + m];                 // channel 3 -> t=1
            bool  rm  = (tt[m] == 0);
            float v0 = s + bn0 + bm0;
            float v1 = s + bn1 + bm1;
            if (rm || cm) { v0 = NEGBIG; v1 = NEGBIG; }
            if (m > n)    { v0 += NEGBIG; v1 += NEGBIG; }
            o0[(size_t)m << 10] = v0;
            o1[(size_t)m << 10] = v1;
        }
    }
}

// ------------------------------- launcher ----------------------------------
extern "C" void kernel_launch(void* const* d_in, const int* in_sizes, int n_in,
                              void* d_out, int out_size, void* d_ws, size_t ws_size,
                              hipStream_t stream) {
    const float* hidden = (const float*)d_in[0];   // (16,1024,1024) f32
    const int*   ttid   = (const int*)  d_in[1];   // (16,1024) i32
    const float* W1     = (const float*)d_in[2];   // (1024,2048) f32
    const float* b1     = (const float*)d_in[3];   // (2048,) f32
    const float* W2     = (const float*)d_in[4];   // (1024,4) f32
    const float* b2     = (const float*)d_in[5];   // (4,) f32
    float*       out    = (float*)d_out;           // (16,2,1024,1024) f32

    char* ws = (char*)d_ws;
    unsigned short* hidbf = (unsigned short*)(ws + 0);          // 32 MB
    unsigned short* w1t   = (unsigned short*)(ws + 33554432);   //  4 MB
    unsigned short* qb    = (unsigned short*)(ws + 37748736);   // 32 MB
    unsigned short* kb    = (unsigned short*)(ws + 71303168);   // 32 MB
    float*          cosT  = (float*)        (ws + 104857600);   //  2 MB
    float*          sinT  = (float*)        (ws + 106954752);   //  2 MB
    float*          bias2 = (float*)        (ws + 109051904);   // 256 KB

    k_cvt_hidden<<<65536, 256, 0, stream>>>(hidden, hidbf);
    k_w1t       <<<8192,  256, 0, stream>>>(W1, w1t);
    k_rope_tab  <<<2048,  256, 0, stream>>>(cosT, sinT);
    k_bias2     <<<64,    256, 0, stream>>>(hidden, W2, b2, bias2);

    // 1024 M-tiles x 32 N-groups = 32768 waves, 8 waves/block
    gemm1_rope<<<4096, 256, 0, stream>>>(hidbf, w1t, b1, cosT, sinT, qb, kb);

    // 16 batches x 64 M-tiles x 16 N-groups = 16384 waves
    gemm2_out<<<2048, 256, 0, stream>>>(qb, kb, bias2, ttid, out);
}
